// QGAAttention_5935644803257
// MI455X (gfx1250) — matrix operand order repaired and checked by reading, not compiled, not run on hardware
//
#include <hip/hip_runtime.h>
#include <hip/hip_bf16.h>

// ---------------------------------------------------------------------------
// GQA attention for MI455X (gfx1250, wave32).
// All matmuls via v_wmma_f32_16x16x32_bf16. Softmax reductions via DPP16 VALU
// (no ds_bpermute), 64-token S-chunks to amortize softmax over 16 WMMAs.
// ---------------------------------------------------------------------------

typedef __bf16 bhalf;
typedef __attribute__((ext_vector_type(16))) __bf16 v16bf;
typedef __attribute__((ext_vector_type(8)))  __bf16 v8bf;
typedef __attribute__((ext_vector_type(4)))  __bf16 v4bf;
typedef __attribute__((ext_vector_type(8)))  float  v8f;

#define HIDDEN   1024
#define NHEADS   16
#define NKV      4
#define HDIM     64
#define SEQ      2048
#define BATCH    4
#define MTOT     (BATCH * SEQ)          // 8192
// softmax uses exp2: scale = log2(e) / sqrt(64); pre-folded into Q.
#define SLOG2    0.18033688011112042f

__device__ inline v16bf cat8(v8bf lo, v8bf hi) {
  return __builtin_shufflevector(lo, hi, 0,1,2,3,4,5,6,7,8,9,10,11,12,13,14,15);
}

__device__ inline v8f wmma_bf16(v16bf a, v16bf b, v8f c) {
  // (neg_a, A, neg_b, B, c_mod, C, reuse_a, reuse_b)
  return __builtin_amdgcn_wmma_f32_16x16x32_bf16(false, a, false, b, (short)0, c,
                                                 false, false);
}

// ---- DPP16 cross-lane helpers (stay within each 16-lane row) --------------
template <int CTRL>
__device__ inline float dpp_mov_f32(float x) {
  return __builtin_bit_cast(
      float, __builtin_amdgcn_mov_dpp(__builtin_bit_cast(int, x), CTRL, 0xf, 0xf,
                                      true));
}
// Reduce across the 16 column lanes of a C-layout tile row (per half-wave).
__device__ inline float row_max16(float x) {
  x = fmaxf(x, dpp_mov_f32<0xB1>(x));    // quad_perm(1,0,3,2)  : xor 1
  x = fmaxf(x, dpp_mov_f32<0x4E>(x));    // quad_perm(2,3,0,1)  : xor 2
  x = fmaxf(x, dpp_mov_f32<0x141>(x));   // row_half_mirror     : xor 4 (reduced)
  x = fmaxf(x, dpp_mov_f32<0x140>(x));   // row_mirror          : xor 8 (reduced)
  return x;
}
__device__ inline float row_sum16(float x) {
  x += dpp_mov_f32<0xB1>(x);
  x += dpp_mov_f32<0x4E>(x);
  x += dpp_mov_f32<0x141>(x);
  x += dpp_mov_f32<0x140>(x);
  return x;
}

// ---------------------------------------------------------------------------
// f32 -> bf16 elementwise convert (x), 4 elems/thread
// ---------------------------------------------------------------------------
__global__ void __launch_bounds__(256)
cvt_f32_bf16_kernel(const float* __restrict__ x, bhalf* __restrict__ y, int n4) {
  int i = blockIdx.x * 256 + threadIdx.x;
  if (i < n4) {
    float4 v = ((const float4*)x)[i];
    v4bf o = { (bhalf)v.x, (bhalf)v.y, (bhalf)v.z, (bhalf)v.w };
    ((v4bf*)y)[i] = o;
  }
}

// ---------------------------------------------------------------------------
// W[k][n] f32  ->  WT[n][k] bf16   (output-coalesced over k)
// ---------------------------------------------------------------------------
__global__ void __launch_bounds__(256)
transpose_w_kernel(const float* __restrict__ W, bhalf* __restrict__ WT,
                   int K, int Nc) {
  int idx = blockIdx.x * 256 + threadIdx.x;
  if (idx >= K * Nc) return;
  int n = idx / K;
  int k = idx - n * K;
  WT[(size_t)n * K + k] = (bhalf)W[(size_t)k * Nc + n];
}

// ---------------------------------------------------------------------------
// WMMA GEMM:  C[M][Nc] = A[M][K](bf16) * BT[Nc][K](bf16)^T + bias
// Block = 256 threads = 8 waves. Block tile 128(M) x 64(N); wave tile 16x64.
// MODE 0: f32 out, row-major (final projection)
// MODE 1: bf16 out scattered to Q layout  [B,H ,N,d], pre-scaled by SLOG2
// MODE 2: bf16 out scattered to K layout  [B,Hkv,S,d]
// MODE 3: bf16 out scattered to V^T layout[B,Hkv,d,S]
// ---------------------------------------------------------------------------
template <int MODE>
__global__ void __launch_bounds__(256)
gemm_wmma_kernel(const bhalf* __restrict__ A, const bhalf* __restrict__ BT,
                 const float* __restrict__ bias, void* __restrict__ Cout,
                 int M, int K, int Nc) {
  const int nTilesN = Nc >> 6;
  const int bm = blockIdx.x / nTilesN;
  const int bn = blockIdx.x - bm * nTilesN;
  const int wave = threadIdx.x >> 5;
  const int lane = threadIdx.x & 31;
  const int mr = lane & 15;     // row (A) / col (B,C) within tile
  const int kh = lane >> 4;     // half-wave selector

  const int m0 = bm * 128 + wave * 16;
  const int n0 = bn * 64;

  v8f acc[4] = {v8f{}, v8f{}, v8f{}, v8f{}};

  const bhalf* arow = A + (size_t)(m0 + mr) * K;
  const bhalf* bcol[4];
#pragma unroll
  for (int c = 0; c < 4; ++c)
    bcol[c] = BT + (size_t)(n0 + 16 * c + mr) * K + kh * 16;

  for (int k0 = 0; k0 < K; k0 += 32) {
    // A fragment: lane holds row m0+mr; 16 contiguous K values (kh*16..+16)
    v8bf alo = *(const v8bf*)(arow + k0 + kh * 16);
    v8bf ahi = *(const v8bf*)(arow + k0 + kh * 16 + 8);
    v16bf a = cat8(alo, ahi);
#pragma unroll
    for (int c = 0; c < 4; ++c) {
      // B fragment: lane holds column, 16 contiguous K values (kh*16..+16)
      v16bf b = *(const v16bf*)(bcol[c] + k0);
      acc[c] = wmma_bf16(a, b, acc[c]);
    }
  }

  // Epilogue. C layout: lane holds col n0+16c+mr; reg r holds row m0+r+8*kh.
#pragma unroll
  for (int c = 0; c < 4; ++c) {
    const int col = n0 + 16 * c + mr;
    const float bv = bias[col];
#pragma unroll
    for (int r = 0; r < 8; ++r) {
      const int row = m0 + r + 8 * kh;
      float val = acc[c][r] + bv;
      if constexpr (MODE == 0) {
        ((float*)Cout)[(size_t)row * Nc + col] = val;
      } else if constexpr (MODE == 1) {
        // Q: [B,16,2048,64], pre-scale by log2(e)/sqrt(d)
        val *= SLOG2;
        const size_t idx = ((((size_t)(row >> 11)) * 16 + (col >> 6)) * SEQ +
                            (row & 2047)) * HDIM + (col & 63);
        ((bhalf*)Cout)[idx] = (bhalf)val;
      } else if constexpr (MODE == 2) {
        // K: [B,4,2048,64]
        const size_t idx = ((((size_t)(row >> 11)) * 4 + (col >> 6)) * SEQ +
                            (row & 2047)) * HDIM + (col & 63);
        ((bhalf*)Cout)[idx] = (bhalf)val;
      } else {
        // V^T: [B,4,64,2048]
        const size_t idx = ((((size_t)(row >> 11)) * 4 + (col >> 6)) * HDIM +
                            (col & 63)) * SEQ + (row & 2047);
        ((bhalf*)Cout)[idx] = (bhalf)val;
      }
    }
  }
}

// ---------------------------------------------------------------------------
// Flash attention: one wave owns a 16(q) x 64(d) tile of one (b, head).
// Streams S in 64-token chunks: 8 score WMMAs, one online-softmax pass with
// DPP16 reductions, P transposed (C-layout -> A-layout) through per-wave LDS,
// then 8 P*V WMMAs (two K=32 steps). Q arrives pre-scaled by log2e/sqrt(d).
// Output to Ob [B, N, 16*64] bf16, head slot = (h%4)*4 + h/4 (ref permutation).
// ---------------------------------------------------------------------------
__global__ void __launch_bounds__(256)
attn_wmma_kernel(const bhalf* __restrict__ Qb, const bhalf* __restrict__ Kb,
                 const bhalf* __restrict__ VbT, bhalf* __restrict__ Ob) {
  __shared__ __align__(16) bhalf lds_p[8][16 * 64];   // 2 KB per wave

  const int wave = threadIdx.x >> 5;
  const int lane = threadIdx.x & 31;
  const int mr = lane & 15;
  const int kh = lane >> 4;

  const int bh = blockIdx.x >> 4;          // 0..63  = b*16 + h
  const int b = bh >> 4;
  const int h = bh & 15;
  const int q0 = ((blockIdx.x & 15) * 8 + wave) * 16;
  const int hkv = h >> 2;                  // kv head = h / GROUPS
  const int ohead = (h & 3) * 4 + hkv;     // output head slot (g*NKV + hkv)

  const bhalf* Qp = Qb + ((((size_t)b * 16 + h) * SEQ) + q0) * HDIM;
  const bhalf* Kp = Kb + (((size_t)b * 4 + hkv) * SEQ) * HDIM;
  const bhalf* Vp = VbT + (((size_t)b * 4 + hkv) * HDIM) * SEQ;
  bhalf* lp = &lds_p[wave][0];

  // Q A-fragments for d = [0,32) and [32,64)
  v16bf aQ[2];
#pragma unroll
  for (int t = 0; t < 2; ++t) {
    const bhalf* qr = Qp + (size_t)mr * HDIM + 32 * t;
    v8bf lo = *(const v8bf*)(qr + kh * 16);
    v8bf hi = *(const v8bf*)(qr + kh * 16 + 8);
    aQ[t] = cat8(lo, hi);
  }

  v8f m_run, l_run;
  v8f acc[4] = {v8f{}, v8f{}, v8f{}, v8f{}};
#pragma unroll
  for (int r = 0; r < 8; ++r) { m_run[r] = -3.0e38f; l_run[r] = 0.0f; }

  for (int s0 = 0; s0 < SEQ; s0 += 64) {
    if (s0 + 64 < SEQ) {  // prefetch next chunk of K rows and V^T rows
      __builtin_prefetch(Kp + (size_t)(s0 + 64 + lane * 2) * HDIM, 0, 1);
      __builtin_prefetch(Vp + (size_t)(lane * 2) * SEQ + s0 + 64, 0, 1);
    }

    // ---- scores: four 16x16 tiles (tokens s0+16t .. +16) ----
    v8f st[4];
#pragma unroll
    for (int t = 0; t < 4; ++t) {
      const int tok = s0 + 16 * t + mr;          // lane = token column
      const bhalf* kr = Kp + (size_t)tok * HDIM;
      v16bf bk0 = *(const v16bf*)(kr + kh * 16);        // d = 0..31
      v16bf bk1 = *(const v16bf*)(kr + 32 + kh * 16);   // d = 32..63
      v8f s = v8f{};
      s = wmma_bf16(aQ[0], bk0, s);
      s = wmma_bf16(aQ[1], bk1, s);
      st[t] = s;   // already scaled: Q carries log2e/sqrt(d)
    }

    // ---- online softmax (rows r+8*kh; DPP reduce over 16 column lanes) ----
    v8f rm;
#pragma unroll
    for (int r = 0; r < 8; ++r) {
      float m01 = fmaxf(st[0][r], st[1][r]);
      float m23 = fmaxf(st[2][r], st[3][r]);
      rm[r] = row_max16(fmaxf(m01, m23));
    }

    v8f m_new, alpha, rs;
#pragma unroll
    for (int r = 0; r < 8; ++r) {
      m_new[r] = fmaxf(m_run[r], rm[r]);
      alpha[r] = __builtin_amdgcn_exp2f(m_run[r] - m_new[r]);
      float p0 = __builtin_amdgcn_exp2f(st[0][r] - m_new[r]);
      float p1 = __builtin_amdgcn_exp2f(st[1][r] - m_new[r]);
      float p2 = __builtin_amdgcn_exp2f(st[2][r] - m_new[r]);
      float p3 = __builtin_amdgcn_exp2f(st[3][r] - m_new[r]);
      st[0][r] = p0; st[1][r] = p1; st[2][r] = p2; st[3][r] = p3;
      rs[r] = row_sum16((p0 + p1) + (p2 + p3));
    }
#pragma unroll
    for (int r = 0; r < 8; ++r) {
      l_run[r] = l_run[r] * alpha[r] + rs[r];
      m_run[r] = m_new[r];
    }
#pragma unroll
    for (int c = 0; c < 4; ++c)
#pragma unroll
      for (int r = 0; r < 8; ++r) acc[c][r] *= alpha[r];

    // ---- transpose P (C layout -> A layout) through this wave's LDS ----
#pragma unroll
    for (int t = 0; t < 4; ++t)
#pragma unroll
      for (int r = 0; r < 8; ++r)
        lp[(r + 8 * kh) * 64 + 16 * t + mr] = (bhalf)st[t][r];

    // ---- P (16x64) * V (64x64): two K=32 steps, B frags from V^T ----
#pragma unroll
    for (int kk = 0; kk < 2; ++kk) {
      v8bf plo = *(const v8bf*)(lp + mr * 64 + 32 * kk + kh * 16);
      v8bf phi = *(const v8bf*)(lp + mr * 64 + 32 * kk + kh * 16 + 8);
      v16bf aP = cat8(plo, phi);
#pragma unroll
      for (int c = 0; c < 4; ++c) {
        const bhalf* vr =
            Vp + (size_t)(16 * c + mr) * SEQ + s0 + 32 * kk + kh * 16;
        v16bf bvf = *(const v16bf*)vr;
        acc[c] = wmma_bf16(aP, bvf, acc[c]);
      }
    }
  }

  // ---- normalize and scatter to Ob [B, N, 1024] bf16 ----
  v8f rl;
#pragma unroll
  for (int r = 0; r < 8; ++r) rl[r] = 1.0f / l_run[r];
  const size_t obase = (size_t)b * SEQ * HIDDEN;
#pragma unroll
  for (int c = 0; c < 4; ++c) {
    const int col = ohead * HDIM + 16 * c + mr;
#pragma unroll
    for (int r = 0; r < 8; ++r) {
      const int tok = q0 + r + 8 * kh;
      Ob[obase + (size_t)tok * HIDDEN + col] = (bhalf)(acc[c][r] * rl[r]);
    }
  }
}

// ---------------------------------------------------------------------------
// launch
// ---------------------------------------------------------------------------
extern "C" void kernel_launch(void* const* d_in, const int* in_sizes, int n_in,
                              void* d_out, int out_size, void* d_ws, size_t ws_size,
                              hipStream_t stream) {
  (void)in_sizes; (void)n_in; (void)out_size; (void)ws_size;
  const float* x  = (const float*)d_in[0];
  const float* Wq = (const float*)d_in[1];
  const float* bq = (const float*)d_in[2];
  const float* Wk = (const float*)d_in[3];
  const float* bk = (const float*)d_in[4];
  const float* Wv = (const float*)d_in[5];
  const float* bv = (const float*)d_in[6];
  const float* Wo = (const float*)d_in[7];
  const float* bo = (const float*)d_in[8];
  float* out = (float*)d_out;

  char* ws = (char*)d_ws;
  size_t o = 0;
  bhalf* xb  = (bhalf*)(ws + o); o += (size_t)MTOT * HIDDEN * 2;       // 16 MB
  bhalf* WqT = (bhalf*)(ws + o); o += (size_t)HIDDEN * HIDDEN * 2;     //  2 MB
  bhalf* WkT = (bhalf*)(ws + o); o += (size_t)256 * HIDDEN * 2;
  bhalf* WvT = (bhalf*)(ws + o); o += (size_t)256 * HIDDEN * 2;
  bhalf* WoT = (bhalf*)(ws + o); o += (size_t)HIDDEN * HIDDEN * 2;
  bhalf* Qb  = (bhalf*)(ws + o); o += (size_t)BATCH * NHEADS * SEQ * HDIM * 2;
  bhalf* Kb  = (bhalf*)(ws + o); o += (size_t)BATCH * NKV * SEQ * HDIM * 2;
  bhalf* VbT = (bhalf*)(ws + o); o += (size_t)BATCH * NKV * HDIM * SEQ * 2;
  bhalf* Ob  = (bhalf*)(ws + o); o += (size_t)MTOT * HIDDEN * 2;

  // 1) convert x to bf16
  {
    int n4 = MTOT * HIDDEN / 4;
    cvt_f32_bf16_kernel<<<n4 / 256, 256, 0, stream>>>(x, xb, n4);
  }
  // 2) transpose + convert weights
  transpose_w_kernel<<<(HIDDEN * HIDDEN) / 256, 256, 0, stream>>>(Wq, WqT, HIDDEN, HIDDEN);
  transpose_w_kernel<<<(HIDDEN * 256) / 256, 256, 0, stream>>>(Wk, WkT, HIDDEN, 256);
  transpose_w_kernel<<<(HIDDEN * 256) / 256, 256, 0, stream>>>(Wv, WvT, HIDDEN, 256);
  transpose_w_kernel<<<(HIDDEN * HIDDEN) / 256, 256, 0, stream>>>(Wo, WoT, HIDDEN, HIDDEN);

  // 3) QKV projections (WMMA, scattered epilogues; Q pre-scaled)
  gemm_wmma_kernel<1><<<(MTOT / 128) * (HIDDEN / 64), 256, 0, stream>>>(
      xb, WqT, bq, Qb, MTOT, HIDDEN, HIDDEN);
  gemm_wmma_kernel<2><<<(MTOT / 128) * (256 / 64), 256, 0, stream>>>(
      xb, WkT, bk, Kb, MTOT, HIDDEN, 256);
  gemm_wmma_kernel<3><<<(MTOT / 128) * (256 / 64), 256, 0, stream>>>(
      xb, WvT, bv, VbT, MTOT, HIDDEN, 256);

  // 4) attention (flash, WMMA): 1024 blocks x 8 waves, 1 q-tile per wave
  attn_wmma_kernel<<<BATCH * NHEADS * (SEQ / 16 / 8), 256, 0, stream>>>(
      Qb, Kb, VbT, Ob);

  // 5) output projection -> f32
  gemm_wmma_kernel<0><<<(MTOT / 128) * (HIDDEN / 64), 256, 0, stream>>>(
      Ob, WoT, bo, out, MTOT, HIDDEN, HIDDEN);
}
